// Attention_86122684220037
// MI455X (gfx1250) — matrix operand order repaired and checked by reading, not compiled, not run on hardware
//
#include <hip/hip_runtime.h>
#include <hip/hip_bf16.h>

typedef __bf16 bf16_t;
typedef __attribute__((ext_vector_type(16))) __bf16 v16bf;
typedef __attribute__((ext_vector_type(8)))  __bf16 v8bf;
typedef __attribute__((ext_vector_type(8)))  float  v8f;
typedef __attribute__((ext_vector_type(4)))  unsigned int v4u;
typedef __attribute__((ext_vector_type(8)))  int v8i;
typedef __attribute__((ext_vector_type(4)))  int v4i;

#if defined(__gfx1250__) && __has_builtin(__builtin_amdgcn_tensor_load_to_lds) && \
    __has_builtin(__builtin_amdgcn_s_wait_tensorcnt)
#define USE_TDM 1
#else
#define USE_TDM 0
#endif

__device__ __forceinline__ v8f wmma_bf16(v16bf a, v16bf b, v8f c) {
  // D = A(16x32 bf16) x B(32x16 bf16) + C(16x16 f32)
  return __builtin_amdgcn_wmma_f32_16x16x32_bf16(
      /*neg_a=*/false, a, /*neg_b=*/false, b,
      /*c_mod=*/(short)0, c, /*reuse_a=*/false, /*reuse_b=*/false);
}

__device__ __forceinline__ float redmax16(float v) {
  v = fmaxf(v, __shfl_xor(v, 1, 16));
  v = fmaxf(v, __shfl_xor(v, 2, 16));
  v = fmaxf(v, __shfl_xor(v, 4, 16));
  v = fmaxf(v, __shfl_xor(v, 8, 16));
  return v;
}
__device__ __forceinline__ float redsum16(float v) {
  v += __shfl_xor(v, 1, 16);
  v += __shfl_xor(v, 2, 16);
  v += __shfl_xor(v, 4, 16);
  v += __shfl_xor(v, 8, 16);
  return v;
}
__device__ __forceinline__ float redsum32(float v) {
  v += __shfl_xor(v, 1, 32);
  v += __shfl_xor(v, 2, 32);
  v += __shfl_xor(v, 4, 32);
  v += __shfl_xor(v, 8, 32);
  v += __shfl_xor(v, 16, 32);
  return v;
}

#if USE_TDM
// Issue a TDM 2D tile load: tile_d1 rows x tile_d0 bf16 elems (row stride =
// d0_stride elems) from `gaddr` into contiguous LDS at byte offset
// `lds_byte_addr`. D# layout per CDNA5 ISA 8.3/8.4 (group0 + group1; 2D ->
// groups 2/3 zero). 6-arg builtin form (clang-23):
//   (uint32x4 g0, int32x8 g1, int32x4 g2, int32x4 g3, int32x8 g4, i32 cpol)
// Wave-level op; EXEC ignored; tracked by TENSORcnt.
__device__ __forceinline__ void tdm_load_tile_2d(unsigned lds_byte_addr,
                                                 const void* gaddr,
                                                 unsigned tile_d0,
                                                 unsigned tile_d1,
                                                 unsigned tensor_d0,
                                                 unsigned tensor_d1,
                                                 unsigned long long d0_stride) {
  const unsigned long long ga = (unsigned long long)gaddr;
  v4u g0;
  g0[0] = 1u;                                        // count=1, user mode
  g0[1] = lds_byte_addr;                             // lds_addr
  g0[2] = (unsigned)(ga & 0xFFFFFFFFu);              // global_addr[31:0]
  g0[3] = (unsigned)((ga >> 32) & 0x01FFFFFFu)       // global_addr[56:32]
          | (2u << 30);                              // type=2 (image)
  v8i g1;
  g1[0] = (int)(1u << 16);                           // data_size=1 (2 bytes)
  g1[1] = (int)((tensor_d0 & 0xFFFFu) << 16);        // tensor_dim0[15:0]
  g1[2] = (int)((tensor_d0 >> 16) | ((tensor_d1 & 0xFFFFu) << 16));
  g1[3] = (int)(((tensor_d1 >> 16) & 0xFFFFu) | (tile_d0 << 16));
  g1[4] = (int)(tile_d1 & 0xFFFFu);                  // tile_dim1; tile_dim2=0
  g1[5] = (int)(d0_stride & 0xFFFFFFFFull);          // tensor_dim0_stride lo
  g1[6] = (int)((d0_stride >> 32) & 0xFFFFull);      // stride hi; dim1_stride=0
  g1[7] = 0;
  const v4i z4 = {};
  const v8i z8 = {};
  __builtin_amdgcn_tensor_load_to_lds(g0, g1, z4, z4, z8, 0);
}
#endif

// ---------------------------------------------------------------------------
// f32 -> bf16 cast
// ---------------------------------------------------------------------------
__global__ void f32_to_bf16_kernel(const float* __restrict__ in,
                                   bf16_t* __restrict__ out, int n) {
  int i = blockIdx.x * blockDim.x + threadIdx.x;
  if (i < n) out[i] = (bf16_t)in[i];
}

// ---------------------------------------------------------------------------
// GEMM: C[M,N] = A[M,K] * W[N,K]^T  (both operands K-major, bf16 in, f32 out)
// Block: 256 threads = 8 waves; block tile 128(M) x 64(N), k-step 32.
// The shared W tile (64 rows x 32 k = 4KB) is TDM-staged into double-buffered
// LDS once per block per k-step, removing the 8x per-wave redundancy.
// ---------------------------------------------------------------------------
__global__ __launch_bounds__(256)
void gemm_bf16_wmma(const bf16_t* __restrict__ A,
                    const bf16_t* __restrict__ W,
                    float* __restrict__ C,
                    int M, int N, int K) {
  __shared__ bf16_t ldsB[2][64 * 32];  // double-buffered B tile

  const int lane = threadIdx.x & 31;
  const int wave = threadIdx.x >> 5;
  const int m16  = lane & 15;
  const int h    = lane >> 4;
  const int rowBase = (blockIdx.y * 8 + wave) * 16;
  const int colBase = blockIdx.x * 64;

  v8f acc[4];
#pragma unroll
  for (int t = 0; t < 4; ++t) acc[t] = (v8f){};

  const bf16_t* arow = A + (size_t)(rowBase + m16) * K;

#if USE_TDM
  const unsigned ldsBase0 = (unsigned)(size_t)(&ldsB[0][0]);
  const unsigned ldsBase1 = (unsigned)(size_t)(&ldsB[1][0]);
  if (wave == 0) {
    tdm_load_tile_2d(ldsBase0, W + (size_t)colBase * K, 32u, 64u,
                     (unsigned)K, (unsigned)N, (unsigned long long)K);
  }
#else
  {  // cooperative stage of buffer 0: 256 threads x 8 bf16 = 64x32 tile
    const int brow = threadIdx.x >> 2;
    const int bcol = (threadIdx.x & 3) * 8;
    *(v8bf*)(&ldsB[0][brow * 32 + bcol]) =
        *(const v8bf*)(W + (size_t)(colBase + brow) * K + bcol);
  }
#endif

  int cur = 0;
  for (int k = 0; k < K; k += 32) {
#if USE_TDM
    if (wave == 0) __builtin_amdgcn_s_wait_tensorcnt(0);
#endif
    __syncthreads();  // buf[cur] ready; previous reads of buf[cur^1] retired

    if (k + 32 < K) {
#if USE_TDM
      if (wave == 0) {
        tdm_load_tile_2d(cur ? ldsBase0 : ldsBase1,
                         W + (size_t)colBase * K + (k + 32), 32u, 64u,
                         (unsigned)K, (unsigned)N, (unsigned long long)K);
      }
#else
      const int brow = threadIdx.x >> 2;
      const int bcol = (threadIdx.x & 3) * 8;
      *(v8bf*)(&ldsB[cur ^ 1][brow * 32 + bcol]) =
          *(const v8bf*)(W + (size_t)(colBase + brow) * K + (k + 32) + bcol);
#endif
    }

    // A tile (A-operand layout: lane m16, half h -> 16 contiguous K at 16h)
    v16bf a;
    {
      a = *(const v16bf*)(arow + k + h * 16);
    }

    // B operand from LDS: lane n holds W-row (col) n, 16 contiguous k at h*16
#pragma unroll
    for (int t = 0; t < 4; ++t) {
      const v16bf b = *(const v16bf*)(&ldsB[cur][(t * 16 + m16) * 32 + h * 16]);
      acc[t] = wmma_bf16(a, b, acc[t]);
    }
    cur ^= 1;
  }

#pragma unroll
  for (int t = 0; t < 4; ++t) {
#pragma unroll
    for (int r = 0; r < 8; ++r) {
      const int row = rowBase + r + 8 * h;
      const int col = colBase + t * 16 + m16;
      C[(size_t)row * N + col] = acc[t][r];
    }
  }
}

// ---------------------------------------------------------------------------
// Fused: RMS-norm(q,k) + RoPE + fold 1/sqrt(hd) into q, cast to bf16,
// relayout q,k -> (B*H, L, 64) and v -> transposed (B*H, 64, L).
// One wave handles one (b,l,h); lane owns elements {lane, lane+32}.
// ---------------------------------------------------------------------------
__global__ __launch_bounds__(256)
void qkv_norm_rope(const float* __restrict__ qkv,   // (B*L, 3*H*64), [s][h][d]
                   const float* __restrict__ cosT,  // (L, 64)
                   const float* __restrict__ sinT,  // (L, 64)
                   const float* __restrict__ qw,    // (64,)
                   const float* __restrict__ kw,    // (64,)
                   bf16_t* __restrict__ Qo,         // (B*H, L, 64)
                   bf16_t* __restrict__ Ko,         // (B*H, L, 64)
                   bf16_t* __restrict__ Vt,         // (B*H, 64, L)
                   int B, int L, int H) {
  const int lane = threadIdx.x & 31;
  const int wave = threadIdx.x >> 5;
  const int gid  = blockIdx.x * 8 + wave;           // (b, l, h) flat
  const int hh   = gid % H;
  const int tmp  = gid / H;
  const int l    = tmp % L;
  const int b    = tmp / L;
  const int C3   = 3 * H * 64;

  const float* base = qkv + (size_t)(b * L + l) * C3 + hh * 64;
  const int e0 = lane, e1 = lane + 32;

  float q0 = base[e0],              q1 = base[e1];
  float k0 = base[H * 64 + e0],     k1 = base[H * 64 + e1];
  float v0 = base[2 * H * 64 + e0], v1 = base[2 * H * 64 + e1];

  const float sq = redsum32(q0 * q0 + q1 * q1);
  const float sk = redsum32(k0 * k0 + k1 * k1);
  const float rq = rsqrtf(sq * (1.0f / 64.0f) + 1e-6f);
  const float rk = rsqrtf(sk * (1.0f / 64.0f) + 1e-6f);
  q0 *= rq * qw[e0]; q1 *= rq * qw[e1];
  k0 *= rk * kw[e0]; k1 *= rk * kw[e1];

  const float c0 = cosT[l * 64 + e0], c1 = cosT[l * 64 + e1];
  const float s0 = sinT[l * 64 + e0], s1 = sinT[l * 64 + e1];
  // rotate_half: out[i] = x[i]*cos[i] - x[i+32]*sin[i]
  //              out[i+32] = x[i+32]*cos[i+32] + x[i]*sin[i+32]
  const float scale = 0.125f;  // 1/sqrt(64), folded into q only
  const float qr0 = (q0 * c0 - q1 * s0) * scale;
  const float qr1 = (q1 * c1 + q0 * s1) * scale;
  const float kr0 = k0 * c0 - k1 * s0;
  const float kr1 = k1 * c1 + k0 * s1;

  const size_t bh = (size_t)b * H + hh;
  bf16_t* qp = Qo + (bh * L + l) * 64;
  bf16_t* kp = Ko + (bh * L + l) * 64;
  qp[e0] = (bf16_t)qr0; qp[e1] = (bf16_t)qr1;
  kp[e0] = (bf16_t)kr0; kp[e1] = (bf16_t)kr1;
  Vt[(bh * 64 + e0) * L + l] = (bf16_t)v0;
  Vt[(bh * 64 + e1) * L + l] = (bf16_t)v1;
}

// ---------------------------------------------------------------------------
// Flash attention: block = 8 waves, each owns 16 query rows of one (b,h);
// iterate keys in steps of 32. K tile (32x64) and V^T tile (64x32) are
// TDM-staged into double-buffered LDS once per block per key-step (all 8
// waves share them), then fed to WMMA as B-operands via ds_load. Online
// softmax in f32 registers; P re-fed as A-operand via per-wave LDS bounce.
// ---------------------------------------------------------------------------
__global__ __launch_bounds__(256)
void attn_flash(const bf16_t* __restrict__ Q,   // (B*H, L, 64), pre-scaled
                const bf16_t* __restrict__ Km,  // (B*H, L, 64)
                const bf16_t* __restrict__ Vt,  // (B*H, 64, L)
                bf16_t* __restrict__ O,         // (B, L, H*64)
                int L, int H) {
  __shared__ bf16_t ldsK[2][32 * 64];  // keys x d, double-buffered (2x4KB)
  __shared__ bf16_t ldsV[2][64 * 32];  // d x keys, double-buffered (2x4KB)
  __shared__ bf16_t ldsP[8][16 * 32];  // per-wave P tile (8KB)

  const int lane = threadIdx.x & 31;
  const int wave = threadIdx.x >> 5;
  const int m16  = lane & 15;
  const int h    = lane >> 4;
  const int qchunks = L / 128;
  const int bh    = blockIdx.x / qchunks;
  const int qBase = (blockIdx.x % qchunks) * 128 + wave * 16;

  const bf16_t* qptr = Q + ((size_t)bh * L + qBase + m16) * 64;
  v16bf qa0, qa1;
  {
    qa0 = *(const v16bf*)(qptr + h * 16);
    qa1 = *(const v16bf*)(qptr + 32 + h * 16);
  }

  float mrow[8], lrow[8];
  v8f o[4];
#pragma unroll
  for (int r = 0; r < 8; ++r) { mrow[r] = -1e30f; lrow[r] = 0.0f; }
#pragma unroll
  for (int t = 0; t < 4; ++t) o[t] = (v8f){};

  const bf16_t* kbase = Km + (size_t)bh * L * 64;
  const bf16_t* vbase = Vt + (size_t)bh * 64 * L;

#if USE_TDM
  const unsigned ldsK0 = (unsigned)(size_t)(&ldsK[0][0]);
  const unsigned ldsK1 = (unsigned)(size_t)(&ldsK[1][0]);
  const unsigned ldsV0 = (unsigned)(size_t)(&ldsV[0][0]);
  const unsigned ldsV1 = (unsigned)(size_t)(&ldsV[1][0]);
  if (wave == 0) {
    // K tile: 32 key rows x 64 d (contiguous rows)
    tdm_load_tile_2d(ldsK0, kbase, 64u, 32u, 64u, (unsigned)L, 64ull);
    // V^T tile: 64 d rows x 32 keys (row stride L)
    tdm_load_tile_2d(ldsV0, vbase, 32u, 64u, (unsigned)L, 64u,
                     (unsigned long long)L);
  }
#else
  {  // cooperative stage of buffer 0
    const int krow = threadIdx.x >> 2;          // 0..63
    const int kcol = (threadIdx.x & 3) * 8;     // 0,8,16,24
    // K: 32 rows x 64 -> 2048 elems; first 128 threads, 16 elems each
    if (threadIdx.x < 128) {
      const int r = threadIdx.x >> 2;
      const int c = (threadIdx.x & 3) * 16;
      *(v16bf*)(&ldsK[0][r * 64 + c]) = *(const v16bf*)(kbase + r * 64 + c);
    }
    *(v8bf*)(&ldsV[0][krow * 32 + kcol]) =
        *(const v8bf*)(vbase + (size_t)krow * L + kcol);
  }
#endif

  int cur = 0;
  for (int kb = 0; kb < L; kb += 32) {
#if USE_TDM
    if (wave == 0) __builtin_amdgcn_s_wait_tensorcnt(0);
#endif
    __syncthreads();  // buf[cur] ready; reads of buf[cur^1] retired

    if (kb + 32 < L) {
#if USE_TDM
      if (wave == 0) {
        tdm_load_tile_2d(cur ? ldsK0 : ldsK1, kbase + (size_t)(kb + 32) * 64,
                         64u, 32u, 64u, (unsigned)L, 64ull);
        tdm_load_tile_2d(cur ? ldsV0 : ldsV1, vbase + (kb + 32),
                         32u, 64u, (unsigned)L, 64u, (unsigned long long)L);
      }
#else
      const int krow = threadIdx.x >> 2;
      const int kcol = (threadIdx.x & 3) * 8;
      if (threadIdx.x < 128) {
        const int r = threadIdx.x >> 2;
        const int c = (threadIdx.x & 3) * 16;
        *(v16bf*)(&ldsK[cur ^ 1][r * 64 + c]) =
            *(const v16bf*)(kbase + (size_t)(kb + 32 + r) * 64 + c);
      }
      *(v8bf*)(&ldsV[cur ^ 1][krow * 32 + kcol]) =
          *(const v8bf*)(vbase + (size_t)krow * L + (kb + 32) + kcol);
#endif
    }

    // ---- S = Q K^T for 32 keys (two 16-key tiles), K from LDS ----
    v8f s[2];
#pragma unroll
    for (int jt = 0; jt < 2; ++jt) {
      s[jt] = (v8f){};
      const bf16_t* kp = &ldsK[cur][(jt * 16 + m16) * 64 + h * 16];
      const v16bf b0 = *(const v16bf*)kp;          // d in [0,32)
      const v16bf b1 = *(const v16bf*)(kp + 32);   // d in [32,64)
      s[jt] = wmma_bf16(qa0, b0, s[jt]);
      s[jt] = wmma_bf16(qa1, b1, s[jt]);
    }

    // ---- online softmax; stage P into per-wave LDS ----
#pragma unroll
    for (int r = 0; r < 8; ++r) {
      float rm = redmax16(fmaxf(s[0][r], s[1][r]));
      const float mn   = fmaxf(mrow[r], rm);
      const float corr = __expf(mrow[r] - mn);
      const float p0   = __expf(s[0][r] - mn);
      const float p1   = __expf(s[1][r] - mn);
      const float rs   = redsum16(p0 + p1);
      lrow[r] = lrow[r] * corr + rs;
      mrow[r] = mn;
#pragma unroll
      for (int t = 0; t < 4; ++t) o[t][r] *= corr;
      const int row = r + 8 * h;
      ldsP[wave][row * 32 + m16]      = (bf16_t)p0;
      ldsP[wave][row * 32 + 16 + m16] = (bf16_t)p1;
    }

    // ---- reload P in A-operand layout (same wave -> DS in-order) ----
    v16bf pa;
#pragma unroll
    for (int i = 0; i < 16; ++i) {
      pa[i] = ldsP[wave][m16 * 32 + h * 16 + i];
    }

    // ---- O += P V, V^T from LDS ----
#pragma unroll
    for (int t = 0; t < 4; ++t) {
      const v16bf vb = *(const v16bf*)(&ldsV[cur][(t * 16 + m16) * 32 + h * 16]);
      o[t] = wmma_bf16(pa, vb, o[t]);
    }
    cur ^= 1;
  }

  const int b  = bh / H;
  const int hh = bh % H;
#pragma unroll
  for (int r = 0; r < 8; ++r) {
    const float rinv = 1.0f / lrow[r];
    const int qrow = qBase + r + 8 * h;
    bf16_t* op = O + ((size_t)(b * L + qrow) * (H * 64)) + hh * 64;
#pragma unroll
    for (int t = 0; t < 4; ++t)
      op[t * 16 + m16] = (bf16_t)(o[t][r] * rinv);
  }
}

// ---------------------------------------------------------------------------
extern "C" void kernel_launch(void* const* d_in, const int* in_sizes, int n_in,
                              void* d_out, int out_size, void* d_ws, size_t ws_size,
                              hipStream_t stream) {
  (void)in_sizes; (void)n_in; (void)out_size; (void)ws_size;
  const float* x        = (const float*)d_in[0];
  const float* cosT     = (const float*)d_in[1];
  const float* sinT     = (const float*)d_in[2];
  const float* w_qkv    = (const float*)d_in[3];
  const float* w_proj   = (const float*)d_in[4];
  const float* q_norm_w = (const float*)d_in[5];
  const float* k_norm_w = (const float*)d_in[6];
  float* out = (float*)d_out;

  const int B = 8, L = 1024, C = 1024, H = 16;
  const int M  = B * L;    // 8192
  const int N1 = 3 * C;    // 3072

  const size_t nX     = (size_t)M * C;
  const size_t nWqkv  = (size_t)N1 * C;
  const size_t nWproj = (size_t)C * C;
  const size_t nQKV   = (size_t)M * N1;
  const size_t nHead  = (size_t)B * H * L * 64;

  char* p = (char*)d_ws;
  bf16_t* x_bf     = (bf16_t*)p; p += nX * 2;
  bf16_t* wqkv_bf  = (bf16_t*)p; p += nWqkv * 2;
  bf16_t* wproj_bf = (bf16_t*)p; p += nWproj * 2;
  float*  qkv_f    = (float*)p;  p += nQKV * 4;
  bf16_t* q_bf     = (bf16_t*)p; p += nHead * 2;
  bf16_t* k_bf     = (bf16_t*)p; p += nHead * 2;
  bf16_t* vt_bf    = (bf16_t*)p; p += nHead * 2;
  bf16_t* attn_bf  = (bf16_t*)p; p += nHead * 2;

  f32_to_bf16_kernel<<<(int)((nX + 255) / 256), 256, 0, stream>>>(x, x_bf, (int)nX);
  f32_to_bf16_kernel<<<(int)((nWqkv + 255) / 256), 256, 0, stream>>>(w_qkv, wqkv_bf, (int)nWqkv);
  f32_to_bf16_kernel<<<(int)((nWproj + 255) / 256), 256, 0, stream>>>(w_proj, wproj_bf, (int)nWproj);

  gemm_bf16_wmma<<<dim3(N1 / 64, M / 128), 256, 0, stream>>>(
      x_bf, wqkv_bf, qkv_f, M, N1, C);

  qkv_norm_rope<<<B * L * H / 8, 256, 0, stream>>>(
      qkv_f, cosT, sinT, q_norm_w, k_norm_w, q_bf, k_bf, vt_bf, B, L, H);

  attn_flash<<<B * H * (L / 128), 256, 0, stream>>>(
      q_bf, k_bf, vt_bf, attn_bf, L, H);

  gemm_bf16_wmma<<<dim3(C / 64, M / 128), 256, 0, stream>>>(
      attn_bf, wproj_bf, out, M, C, C);
}
